// MixtureOfExperts_61821759259057
// MI455X (gfx1250) — matrix operand order repaired and checked by reading, not compiled, not run on hardware
//
#include <hip/hip_runtime.h>
#include <hip/hip_bf16.h>
#include <math.h>

// Problem constants (from reference): B=2,S=2048,H=1024,I=2048,E=8,K=2
#define TOK   4096          // B*S
#define HDIM  1024
#define IDIM  2048
#define NEXP  8
#define TOPK  2
#define CAP   4096          // per-expert list capacity (worst case: all tokens)
#define NASSIGN (TOK*TOPK)  // 8192 total assignments
#define OUTELEMS (TOK*HDIM) // 4194304 ; aux loss at d_out[OUTELEMS]

typedef __attribute__((ext_vector_type(16))) __bf16        v16bf;
typedef __attribute__((ext_vector_type(8)))  float         v8f;
typedef __attribute__((ext_vector_type(8)))  unsigned int  v8u;

// ---------------- workspace layout (bytes) ----------------
// 0        : counts   int[8]
// 64       : offsets  int[8]
// 128      : partials float[16*8]
// 1024     : tke      int[NASSIGN]
// 33792    : tkpos    int[NASSIGN]
// 66560    : tkw      float[NASSIGN]
// 99328    : list     int[NEXP*CAP]
// 230400   : hidden   ushort(bf16)[NASSIGN*IDIM]   = 32 MB
// 33784832 : expert_out float[NASSIGN*HDIM]        = 32 MB
// total ~ 67,339,264 bytes (~64.3 MB) required of d_ws
#define WS_COUNTS   0
#define WS_OFFSETS  64
#define WS_PARTIALS 128
#define WS_TKE      1024
#define WS_TKPOS    33792
#define WS_TKW      66560
#define WS_LIST     99328
#define WS_HIDDEN   230400
#define WS_EOUT     33784832

__device__ __forceinline__ unsigned short f2bf(float f) {
  unsigned int u = __float_as_uint(f);
  unsigned int r = u + 0x7FFFu + ((u >> 16) & 1u);   // round-to-nearest-even
  return (unsigned short)(r >> 16);
}

// Build a v16bf WMMA fragment from two 16-byte LDS reads (ds_load_b128 x2).
__device__ __forceinline__ v16bf ld_frag(const unsigned short* p0,
                                         const unsigned short* p1) {
  uint4 q0 = *(const uint4*)p0;
  uint4 q1 = *(const uint4*)p1;
  v8u pk;
  pk[0] = q0.x; pk[1] = q0.y; pk[2] = q0.z; pk[3] = q0.w;
  pk[4] = q1.x; pk[5] = q1.y; pk[6] = q1.z; pk[7] = q1.w;
  return __builtin_bit_cast(v16bf, pk);
}

// ---------------- kernel 0: zero counts ----------------
__global__ void moe_zero_counts(int* counts) {
  if (threadIdx.x < NEXP) counts[threadIdx.x] = 0;
}

// ---------------- kernel 1: router ----------------
__global__ __launch_bounds__(256) void moe_router(
    const float* __restrict__ x, const float* __restrict__ Wr,
    int* counts, int* list, int* tke, int* tkpos, float* tkw,
    float* partials) {
  const int tid = threadIdx.x;
  const int token = blockIdx.x * 256 + tid;

  float acc[NEXP];
#pragma unroll
  for (int e = 0; e < NEXP; ++e) acc[e] = 0.f;

  const float* xr = x + (size_t)token * HDIM;
  for (int h = 0; h < HDIM; ++h) {
    float xv = xr[h];
    const float* wr = Wr + h * NEXP;
#pragma unroll
    for (int e = 0; e < NEXP; ++e) acc[e] += xv * wr[e];
  }
  float m = acc[0];
#pragma unroll
  for (int e = 1; e < NEXP; ++e) m = fmaxf(m, acc[e]);
  float p[NEXP]; float s = 0.f;
#pragma unroll
  for (int e = 0; e < NEXP; ++e) { p[e] = __expf(acc[e] - m); s += p[e]; }
  float inv = 1.f / s;
#pragma unroll
  for (int e = 0; e < NEXP; ++e) p[e] *= inv;

  int i1 = 0; float p1 = p[0];
#pragma unroll
  for (int e = 1; e < NEXP; ++e) if (p[e] > p1) { p1 = p[e]; i1 = e; }
  int i2 = (i1 == 0) ? 1 : 0; float p2 = p[i2];
#pragma unroll
  for (int e = 0; e < NEXP; ++e)
    if (e != i1 && p[e] > p2) { p2 = p[e]; i2 = e; }
  float rs = 1.f / (p1 + p2);
  float wA = p1 * rs, wB = p2 * rs;

  int posA = atomicAdd(&counts[i1], 1);
  int posB = atomicAdd(&counts[i2], 1);
  list[i1 * CAP + posA] = token;
  list[i2 * CAP + posB] = token;
  tke[token * 2 + 0] = i1;  tkpos[token * 2 + 0] = posA;  tkw[token * 2 + 0] = wA;
  tke[token * 2 + 1] = i2;  tkpos[token * 2 + 1] = posB;  tkw[token * 2 + 1] = wB;

  __shared__ float sred[256];
  for (int e = 0; e < NEXP; ++e) {
    sred[tid] = p[e];
    __syncthreads();
    for (int s2 = 128; s2 > 0; s2 >>= 1) {
      if (tid < s2) sred[tid] += sred[tid + s2];
      __syncthreads();
    }
    if (tid == 0) partials[blockIdx.x * NEXP + e] = sred[0];
    __syncthreads();
  }
}

// ---------------- kernel 2: scan + aux loss ----------------
__global__ void moe_scan(const int* counts, int* offsets,
                         const float* partials, float* d_out) {
  if (threadIdx.x == 0 && blockIdx.x == 0) {
    int off = 0;
    for (int e = 0; e < NEXP; ++e) { offsets[e] = off; off += counts[e]; }
    float aux = 0.f;
    for (int e = 0; e < NEXP; ++e) {
      float us = 0.f;
      for (int b = 0; b < TOK / 256; ++b) us += partials[b * NEXP + e];
      float d = us / (float)TOK - 1.0f / (float)NEXP;
      aux += d * d;
    }
    d_out[OUTELEMS] = aux * 0.01f;
  }
}

// ---------------- kernel 3: gate/up GEMM + SwiGLU (WMMA) ----------------
// Tile: 64 tokens x 128 I-cols, K=HDIM in chunks of 64 (2 WMMA k-steps/stage).
// 8 waves: wave = (cpair<<2)|mrow; each wave owns 16 rows x 64 cols
// (4 subtiles) for BOTH gate and up -> 16 v_wmma per staged chunk.
__global__ __launch_bounds__(256) void moe_gemm1(
    const float* __restrict__ x, const float* __restrict__ w1,
    const float* __restrict__ w2, const int* __restrict__ counts,
    const int* __restrict__ offsets, const int* __restrict__ list,
    unsigned short* __restrict__ hidden) {
  const int e  = blockIdx.z;
  const int m0 = blockIdx.y * 64;
  const int i0 = blockIdx.x * 128;
  const int cnt = counts[e];
  if (m0 >= cnt) return;

  __shared__ unsigned short lX [64][64];   //  8 KB
  __shared__ unsigned short lW1[128][64];  // 16 KB
  __shared__ unsigned short lW2[128][64];  // 16 KB
  __shared__ int rowTok[64];

  const int tid = threadIdx.x;
  if (tid < 64) {
    int r = m0 + tid;
    rowTok[tid] = (r < cnt) ? list[e * CAP + r] : -1;
  }
  const int wave = tid >> 5, lane = tid & 31;
  const int mrow = wave & 3, cpair = wave >> 2;
  const int half = lane >> 4, ln = lane & 15;

  v8f accG[4] = {}; v8f accU[4] = {};
  const float* w1e = w1 + (size_t)e * HDIM * IDIM;
  const float* w2e = w2 + (size_t)e * HDIM * IDIM;

  __syncthreads();
  for (int k0 = 0; k0 < HDIM; k0 += 64) {
    __syncthreads();
    // stage A: gathered x rows, float4 global loads -> 4x bf16 -> uint2 LDS
    for (int idx = tid; idx < 64 * 16; idx += 256) {
      int r = idx >> 4, c4 = (idx & 15) * 4;
      int tok = rowTok[r];
      float4 v = make_float4(0.f, 0.f, 0.f, 0.f);
      if (tok >= 0)
        v = *(const float4*)(x + (size_t)tok * HDIM + k0 + c4);
      uint2 pk;
      pk.x = (unsigned)f2bf(v.x) | ((unsigned)f2bf(v.y) << 16);
      pk.y = (unsigned)f2bf(v.z) | ((unsigned)f2bf(v.w) << 16);
      *(uint2*)&lX[r][c4] = pk;
    }
    // stage B: w1/w2 [64 x 128], float2 loads, transposed into [n][k]
    for (int idx = tid; idx < 64 * 64; idx += 256) {
      int kk = idx >> 6, n2 = (idx & 63) * 2;
      size_t go = (size_t)(k0 + kk) * IDIM + i0 + n2;
      float2 a1 = *(const float2*)(w1e + go);
      float2 a2 = *(const float2*)(w2e + go);
      lW1[n2][kk]     = f2bf(a1.x);
      lW1[n2 + 1][kk] = f2bf(a1.y);
      lW2[n2][kk]     = f2bf(a2.x);
      lW2[n2 + 1][kk] = f2bf(a2.y);
    }
    // prefetch next K-chunk of weights (global_prefetch_b8)
    if (k0 + 64 < HDIM) {
      size_t pgo = (size_t)(k0 + 64 + (tid >> 3)) * IDIM + i0 + (tid & 7) * 16;
      __builtin_prefetch(w1e + pgo, 0, 1);
      __builtin_prefetch(w2e + pgo, 0, 1);
    }
    __syncthreads();

#pragma unroll
    for (int ks = 0; ks < 64; ks += 32) {
      // Load ALL fragments first, then issue back-to-back WMMAs.
      const unsigned short* ar = &lX[mrow * 16 + ln][ks];
      v16bf a = ld_frag(ar + half * 8, ar + 16 + half * 8);
      v16bf b1f[4], b2f[4];
#pragma unroll
      for (int s = 0; s < 4; ++s) {
        int n0 = cpair * 64 + s * 16;
        const unsigned short* b1r = &lW1[n0 + ln][ks];
        const unsigned short* b2r = &lW2[n0 + ln][ks];
        b1f[s] = ld_frag(b1r + half * 8, b1r + 16 + half * 8);
        b2f[s] = ld_frag(b2r + half * 8, b2r + 16 + half * 8);
      }
#pragma unroll
      for (int s = 0; s < 4; ++s) {
        accG[s] = __builtin_amdgcn_wmma_f32_16x16x32_bf16(
            false, a, false, b1f[s], (short)0, accG[s], false, false);
        accU[s] = __builtin_amdgcn_wmma_f32_16x16x32_bf16(
            false, a, false, b2f[s], (short)0, accU[s], false, false);
      }
    }
  }

  // epilogue: hidden = silu(gate) * up -> bf16, slot-indexed rows
  const int base = offsets[e] + m0;
#pragma unroll
  for (int s = 0; s < 4; ++s) {
    int col = i0 + cpair * 64 + s * 16 + ln;
#pragma unroll
    for (int v = 0; v < 8; ++v) {
      int r = mrow * 16 + v + half * 8;            // C/D layout: M = v + half*8
      if (m0 + r < cnt) {
        float g = accG[s][v], u = accU[s][v];
        float hsw = (g / (1.f + __expf(-g))) * u;
        hidden[(size_t)(base + r) * IDIM + col] = f2bf(hsw);
      }
    }
  }
}

// ---------------- kernel 4: down-proj GEMM (WMMA) ----------------
// Tile: 64 rows x 128 H-cols, K=IDIM in chunks of 64.
__global__ __launch_bounds__(256) void moe_gemm2(
    const unsigned short* __restrict__ hidden, const float* __restrict__ w3,
    const int* __restrict__ counts, const int* __restrict__ offsets,
    float* __restrict__ eout) {
  const int e  = blockIdx.z;
  const int m0 = blockIdx.y * 64;
  const int h0 = blockIdx.x * 128;
  const int cnt = counts[e];
  if (m0 >= cnt) return;
  const int base = offsets[e] + m0;

  __shared__ unsigned short lA[64][64];    //  8 KB
  __shared__ unsigned short lB[128][64];   // 16 KB

  const int tid = threadIdx.x;
  const int wave = tid >> 5, lane = tid & 31;
  const int mrow = wave & 3, cpair = wave >> 2;
  const int half = lane >> 4, ln = lane & 15;

  v8f acc[4] = {};
  const float* w3e = w3 + (size_t)e * IDIM * HDIM;

  for (int k0 = 0; k0 < IDIM; k0 += 64) {
    __syncthreads();
    // stage A: hidden already bf16; uint2 (4 elems) copies
    for (int idx = tid; idx < 64 * 16; idx += 256) {
      int r = idx >> 4, c4 = (idx & 15) * 4;
      uint2 v = make_uint2(0u, 0u);
      if (m0 + r < cnt)
        v = *(const uint2*)(hidden + (size_t)(base + r) * IDIM + k0 + c4);
      *(uint2*)&lA[r][c4] = v;
    }
    // stage B: w3 [64 x 128] f32 -> bf16, transposed into [n][k]
    for (int idx = tid; idx < 64 * 64; idx += 256) {
      int kk = idx >> 6, n2 = (idx & 63) * 2;
      float2 a3 = *(const float2*)(w3e + (size_t)(k0 + kk) * HDIM + h0 + n2);
      lB[n2][kk]     = f2bf(a3.x);
      lB[n2 + 1][kk] = f2bf(a3.y);
    }
    if (k0 + 64 < IDIM) {
      size_t pgo = (size_t)(k0 + 64 + (tid >> 3)) * HDIM + h0 + (tid & 7) * 16;
      __builtin_prefetch(w3e + pgo, 0, 1);
    }
    __syncthreads();

#pragma unroll
    for (int ks = 0; ks < 64; ks += 32) {
      const unsigned short* ar = &lA[mrow * 16 + ln][ks];
      v16bf a = ld_frag(ar + half * 8, ar + 16 + half * 8);
      v16bf bf[4];
#pragma unroll
      for (int s = 0; s < 4; ++s) {
        int n0 = cpair * 64 + s * 16;
        const unsigned short* br = &lB[n0 + ln][ks];
        bf[s] = ld_frag(br + half * 8, br + 16 + half * 8);
      }
#pragma unroll
      for (int s = 0; s < 4; ++s)
        acc[s] = __builtin_amdgcn_wmma_f32_16x16x32_bf16(
            false, a, false, bf[s], (short)0, acc[s], false, false);
    }
  }

#pragma unroll
  for (int s = 0; s < 4; ++s) {
    int col = h0 + cpair * 64 + s * 16 + ln;
#pragma unroll
    for (int v = 0; v < 8; ++v) {
      int r = mrow * 16 + v + half * 8;
      if (m0 + r < cnt)
        eout[(size_t)(base + r) * HDIM + col] = acc[s][v];
    }
  }
}

// ---------------- kernel 5: combine ----------------
__global__ __launch_bounds__(256) void moe_combine(
    const float* __restrict__ eout, const int* __restrict__ tke,
    const int* __restrict__ tkpos, const float* __restrict__ tkw,
    const int* __restrict__ offsets, float* __restrict__ out) {
  int idx = blockIdx.x * 256 + threadIdx.x;   // over TOK*HDIM
  int token = idx >> 10;                      // HDIM = 1024
  int h = idx & (HDIM - 1);
  int e0 = tke[token * 2 + 0], e1 = tke[token * 2 + 1];
  int s0 = offsets[e0] + tkpos[token * 2 + 0];
  int s1 = offsets[e1] + tkpos[token * 2 + 1];
  float w0 = tkw[token * 2 + 0], w1 = tkw[token * 2 + 1];
  out[idx] = w0 * eout[(size_t)s0 * HDIM + h] + w1 * eout[(size_t)s1 * HDIM + h];
}

// ---------------- host launcher ----------------
extern "C" void kernel_launch(void* const* d_in, const int* in_sizes, int n_in,
                              void* d_out, int out_size, void* d_ws, size_t ws_size,
                              hipStream_t stream) {
  const float* x  = (const float*)d_in[0];
  const float* Wr = (const float*)d_in[1];
  const float* w1 = (const float*)d_in[2];
  const float* w2 = (const float*)d_in[3];
  const float* w3 = (const float*)d_in[4];
  float* out = (float*)d_out;

  char* ws = (char*)d_ws;
  int*   counts   = (int*)  (ws + WS_COUNTS);
  int*   offsets  = (int*)  (ws + WS_OFFSETS);
  float* partials = (float*)(ws + WS_PARTIALS);
  int*   tke      = (int*)  (ws + WS_TKE);
  int*   tkpos    = (int*)  (ws + WS_TKPOS);
  float* tkw      = (float*)(ws + WS_TKW);
  int*   list     = (int*)  (ws + WS_LIST);
  unsigned short* hidden = (unsigned short*)(ws + WS_HIDDEN);
  float* eout     = (float*)(ws + WS_EOUT);

  moe_zero_counts<<<1, 64, 0, stream>>>(counts);
  moe_router<<<TOK / 256, 256, 0, stream>>>(x, Wr, counts, list, tke, tkpos,
                                            tkw, partials);
  moe_scan<<<1, 32, 0, stream>>>(counts, offsets, partials, out);
  moe_gemm1<<<dim3(IDIM / 128, CAP / 64, NEXP), 256, 0, stream>>>(
      x, w1, w2, counts, offsets, list, hidden);
  moe_gemm2<<<dim3(HDIM / 128, CAP / 64, NEXP), 256, 0, stream>>>(
      hidden, w3, counts, offsets, eout);
  moe_combine<<<(TOK * HDIM) / 256, 256, 0, stream>>>(eout, tke, tkpos, tkw,
                                                      offsets, out);
}
